// MHA_52965536694823
// MI455X (gfx1250) — matrix operand order repaired and checked by reading, not compiled, not run on hardware
//
#include <hip/hip_runtime.h>
#include <hip/hip_bf16.h>

typedef __attribute__((ext_vector_type(16))) _Float16 v16h;
typedef __attribute__((ext_vector_type(8)))  float    v8f;

#define DIM       4096
#define NH        32
#define NKV       8
#define HD        128
#define KVSTRIDE  (NKV*HD)     // 1024 floats per cache row
#define CACHE_LEN 8192
#define TCHUNK    128
#define NCHUNK    64           // 8192 / 128
#define SCALE     0.08838834764831845f   // 1/sqrt(128)

// workspace layout (float offsets)
#define WS_Q    0              // 4096  rotated Q
#define WS_K    4096           // 1024  rotated new K
#define WS_V    5120           // 1024  new V
#define WS_ATTN 6144           // 4096  attention output vector
#define WS_PM   10240          // 2048  partial max   [qh=32][chunk=64]
#define WS_PL   12288          // 2048  partial sumexp
#define WS_PACC 14336          // 262144 partial acc  [qh=32][chunk=64][d=128]

// ---------------------------------------------------------------------------
// Kernel 1: fused QKV GEMV (6144 rows) + RoPE epilogue.
// Block = 256 threads = 8 waves; wave w computes row 8*blockIdx.x + w.
// x (16KB) staged in LDS once per block; weight rows read with b128 loads.
// ---------------------------------------------------------------------------
__global__ __launch_bounds__(256) void qkv_rope_kernel(
    const float* __restrict__ x,  const float* __restrict__ wq,
    const float* __restrict__ wk, const float* __restrict__ wv,
    const int* __restrict__ start_ptr, float* __restrict__ ws)
{
    __shared__ float lds_x[DIM];
    __shared__ float lds_row[8];
    const int t = threadIdx.x;

    for (int i = 0; i < 4; ++i) {
        int idx = (t + i * 256) * 4;
        *(float4*)&lds_x[idx] = *(const float4*)&x[idx];
    }
    __syncthreads();

    const int wave = t >> 5, lane = t & 31;
    const int row = blockIdx.x * 8 + wave;
    const float* wrow;
    if      (row < 4096) wrow = wq + (size_t)row * DIM;
    else if (row < 5120) wrow = wk + (size_t)(row - 4096) * DIM;
    else                 wrow = wv + (size_t)(row - 5120) * DIM;

    float acc = 0.f;
    for (int i = 0; i < 32; ++i) {
        int off = (i * 32 + lane) * 4;
        __builtin_prefetch(&wrow[off + 128], 0, 1);   // one tile ahead
        float4 w4 = *(const float4*)&wrow[off];
        float4 x4 = *(const float4*)&lds_x[off];
        acc += w4.x * x4.x + w4.y * x4.y + w4.z * x4.z + w4.w * x4.w;
    }
    for (int off = 16; off > 0; off >>= 1) acc += __shfl_xor(acc, off, 32);
    if (lane == 0) lds_row[wave] = acc;
    __syncthreads();

    // RoPE: rows come in (even, odd) pairs inside one block (8 | 128).
    if (t < 8) {
        int r = blockIdx.x * 8 + t;
        float self = lds_row[t];
        if (r < 5120) {                       // Q or K rows get RoPE
            int d = r & 127;
            int j = d >> 1;
            float other = lds_row[t ^ 1];
            double inv = pow(500000.0, -(double)(2 * j) / 128.0);
            double ang = fmod((double)(*start_ptr) * inv, 6.283185307179586);
            float c = (float)cos(ang), s = (float)sin(ang);
            float val = (d & 1) ? (other * s + self * c)
                                : (self * c - other * s);
            if (r < 4096) ws[WS_Q + r] = val;
            else          ws[WS_K + (r - 4096)] = val;
        } else {
            ws[WS_V + (r - 5120)] = self;     // V: no RoPE
        }
    }
}

// ---------------------------------------------------------------------------
// Kernel 2: flash-decoding partials. grid = (64 chunks, 8 kv heads).
// Phase 1: scores via v_wmma_f32_16x16x32_f16 (A = 16 keys x 32 dims from
//          K cache, B = 4-head Q group padded to 16 cols in LDS).
// Phase 2: per-chunk online-softmax stats.
// Phase 3: P*V via v_wmma_f32_16x16x32_f16 (A = probs rows, B = V columns),
//          one 16-dim output tile per wave, 4 accumulating k-steps.
// ---------------------------------------------------------------------------
__global__ __launch_bounds__(256) void attn_partial_kernel(
    const float* __restrict__ k_cache, const float* __restrict__ v_cache,
    const int* __restrict__ start_ptr, float* __restrict__ ws)
{
    __shared__ alignas(32) _Float16 lds_q[16 * HD];   // 4KB, rows >=4 zeroed
    __shared__ float lds_s[4 * TCHUNK];               // scores -> exp(scores)
    __shared__ float lds_m[4], lds_l[4];

    const int t = threadIdx.x;
    const int c = blockIdx.x;      // key chunk
    const int h = blockIdx.y;      // kv head
    const int start = *start_ptr;  // 8191: position of the new K/V
    const float* q    = ws + WS_Q;
    const float* knew = ws + WS_K;
    const float* vnew = ws + WS_V;

    // stage Q group (4 heads) as f16 into 16x128 LDS tile, pad rows with 0
    for (int i = 0; i < 8; ++i) {
        int idx = t + i * 256;                 // 0..2047
        int r = idx >> 7, d = idx & 127;
        float v = (r < 4) ? q[(h * 4 + r) * HD + d] : 0.f;
        lds_q[idx] = (_Float16)v;
    }
    __syncthreads();

    const int wave = t >> 5, lane = t & 31;
    const int g = lane >> 4, n = lane & 15;

    // ---- phase 1: WMMA scores, 16 keys per wave --------------------------
    // A as loaded here (contiguous k order, A and B alike): lane holds row M=lane%16; element e holds
    // K = e + 16*(lane/16).
    const int key = c * TCHUNK + wave * 16 + n;
    const float* krow = (key == start) ? (knew + h * HD)
                                       : (k_cache + (size_t)key * KVSTRIDE + h * HD);
    v8f sacc = {};
    for (int ks = 0; ks < 4; ++ks) {
        int d0 = g * 16 + ks * 32;
        float4 a0 = *(const float4*)&krow[d0];
        float4 a1 = *(const float4*)&krow[d0 + 4];
        float4 b0 = *(const float4*)&krow[d0 + 8];
        float4 b1 = *(const float4*)&krow[d0 + 12];
        v16h A;
        A[0]=(_Float16)a0.x;  A[1]=(_Float16)a0.y;  A[2]=(_Float16)a0.z;  A[3]=(_Float16)a0.w;
        A[4]=(_Float16)a1.x;  A[5]=(_Float16)a1.y;  A[6]=(_Float16)a1.z;  A[7]=(_Float16)a1.w;
        A[8]=(_Float16)b0.x;  A[9]=(_Float16)b0.y;  A[10]=(_Float16)b0.z; A[11]=(_Float16)b0.w;
        A[12]=(_Float16)b1.x; A[13]=(_Float16)b1.y; A[14]=(_Float16)b1.z; A[15]=(_Float16)b1.w;
        // B as loaded here (same contiguous k order as A): lane holds col N=lane%16; element e
        // holds K = e + 16*(lane/16). K runs over head dims -> contiguous LDS.
        v16h B = *(const v16h*)&lds_q[n * HD + g * 16 + ks * 32];
        sacc = __builtin_amdgcn_wmma_f32_16x16x32_f16(
                   false, A, false, B, (short)0, sacc, false, false);
    }
    // D layout: VGPR r -> M = r + 8*(lane/16), N = lane%16 (= q head)
    if (n < 4) {
        for (int r = 0; r < 8; ++r) {
            int kk = wave * 16 + r + 8 * g;
            lds_s[n * TCHUNK + kk] = sacc[r] * SCALE;
        }
    }
    __syncthreads();

    // ---- phase 2: per-chunk softmax stats (wave w handles q head w) ------
    if (wave < 4) {
        const int qh = wave;
        float v0 = lds_s[qh * TCHUNK + lane];
        float v1 = lds_s[qh * TCHUNK + lane + 32];
        float v2 = lds_s[qh * TCHUNK + lane + 64];
        float v3 = lds_s[qh * TCHUNK + lane + 96];
        float m = fmaxf(fmaxf(v0, v1), fmaxf(v2, v3));
        for (int off = 16; off > 0; off >>= 1) m = fmaxf(m, __shfl_xor(m, off, 32));
        float e0 = __expf(v0 - m), e1 = __expf(v1 - m);
        float e2 = __expf(v2 - m), e3 = __expf(v3 - m);
        lds_s[qh * TCHUNK + lane]      = e0;
        lds_s[qh * TCHUNK + lane + 32] = e1;
        lds_s[qh * TCHUNK + lane + 64] = e2;
        lds_s[qh * TCHUNK + lane + 96] = e3;
        float ssum = e0 + e1 + e2 + e3;
        for (int off = 16; off > 0; off >>= 1) ssum += __shfl_xor(ssum, off, 32);
        if (lane == 0) { lds_m[qh] = m; lds_l[qh] = ssum; }
    }
    __syncthreads();

    // ---- phase 3: P*V via WMMA. Wave w -> output dims [16w, 16w+16) ------
    // A = probs: row M = q head (rows >=4 zero), K = key in chunk.
    // B = V:     col N = dim within tile, K = key in chunk (lanes 0-15 read
    //            16 consecutive dims of one cache row -> 64B sectors).
    {
        const int dim = 16 * wave + n;
        v8f pv = {};
        for (int ks = 0; ks < 4; ++ks) {
            int kbase = 8 * g + 32 * ks;
            float4 p0 = {0,0,0,0}, p1 = {0,0,0,0}, p2 = {0,0,0,0}, p3 = {0,0,0,0};
            if (n < 4) {
                p0 = *(const float4*)&lds_s[n * TCHUNK + kbase];
                p1 = *(const float4*)&lds_s[n * TCHUNK + kbase + 4];
                p2 = *(const float4*)&lds_s[n * TCHUNK + kbase + 16];
                p3 = *(const float4*)&lds_s[n * TCHUNK + kbase + 20];
            }
            v16h A;
            A[0]=(_Float16)p0.x;  A[1]=(_Float16)p0.y;  A[2]=(_Float16)p0.z;  A[3]=(_Float16)p0.w;
            A[4]=(_Float16)p1.x;  A[5]=(_Float16)p1.y;  A[6]=(_Float16)p1.z;  A[7]=(_Float16)p1.w;
            A[8]=(_Float16)p2.x;  A[9]=(_Float16)p2.y;  A[10]=(_Float16)p2.z; A[11]=(_Float16)p2.w;
            A[12]=(_Float16)p3.x; A[13]=(_Float16)p3.y; A[14]=(_Float16)p3.z; A[15]=(_Float16)p3.w;
            v16h B;
            #pragma unroll
            for (int e = 0; e < 16; ++e) {
                int kg = c * TCHUNK + (e & 7) + 8 * g + 16 * (e >> 3) + 32 * ks;
                const float* vp = (kg == start) ? (vnew + h * HD)
                                                : (v_cache + (size_t)kg * KVSTRIDE + h * HD);
                B[e] = (_Float16)vp[dim];
            }
            pv = __builtin_amdgcn_wmma_f32_16x16x32_f16(
                     false, A, false, B, (short)0, pv, false, false);
        }
        // D: VGPR r -> row M = r + 8*g, col N = n. Valid q heads are M 0..3.
        if (g == 0) {
            float* pacc = ws + WS_PACC;
            for (int r = 0; r < 4; ++r)
                pacc[((size_t)(h * 4 + r) * NCHUNK + c) * HD + dim] = pv[r];
        }
    }
    if (t < 4) {
        ws[WS_PM + (h * 4 + t) * NCHUNK + c] = lds_m[t];
        ws[WS_PL + (h * 4 + t) * NCHUNK + c] = lds_l[t];
    }
}

// ---------------------------------------------------------------------------
// Kernel 3: combine flash-decoding partials. 32 blocks (one per q head).
// ---------------------------------------------------------------------------
__global__ __launch_bounds__(128) void combine_kernel(float* __restrict__ ws)
{
    const int qh = blockIdx.x, d = threadIdx.x;
    const float* pm   = ws + WS_PM + qh * NCHUNK;
    const float* pl   = ws + WS_PL + qh * NCHUNK;
    const float* pacc = ws + WS_PACC + (size_t)qh * NCHUNK * HD;
    float M = -3.4e38f;
    for (int c = 0; c < NCHUNK; ++c) M = fmaxf(M, pm[c]);
    float L = 0.f, o = 0.f;
    for (int c = 0; c < NCHUNK; ++c) {
        float w = __expf(pm[c] - M);
        L += pl[c] * w;
        o += pacc[c * HD + d] * w;
    }
    ws[WS_ATTN + qh * HD + d] = o / L;
}

// ---------------------------------------------------------------------------
// Kernel 4: output projection GEMV: out[r] = wo[r,:] . attn  (4096 rows)
// ---------------------------------------------------------------------------
__global__ __launch_bounds__(256) void oproj_kernel(
    const float* __restrict__ wo, const float* __restrict__ ws,
    float* __restrict__ out)
{
    __shared__ float lds_a[DIM];
    const int t = threadIdx.x;
    const float* attn = ws + WS_ATTN;
    for (int i = 0; i < 4; ++i) {
        int idx = (t + i * 256) * 4;
        *(float4*)&lds_a[idx] = *(const float4*)&attn[idx];
    }
    __syncthreads();

    const int wave = t >> 5, lane = t & 31;
    const int row = blockIdx.x * 8 + wave;
    const float* wrow = wo + (size_t)row * DIM;
    float acc = 0.f;
    for (int i = 0; i < 32; ++i) {
        int off = (i * 32 + lane) * 4;
        __builtin_prefetch(&wrow[off + 128], 0, 1);   // one tile ahead
        float4 w4 = *(const float4*)&wrow[off];
        float4 a4 = *(const float4*)&lds_a[off];
        acc += w4.x * a4.x + w4.y * a4.y + w4.z * a4.z + w4.w * a4.w;
    }
    for (int off = 16; off > 0; off >>= 1) acc += __shfl_xor(acc, off, 32);
    if (lane == 0) out[row] = acc;
}

// ---------------------------------------------------------------------------
extern "C" void kernel_launch(void* const* d_in, const int* in_sizes, int n_in,
                              void* d_out, int out_size, void* d_ws, size_t ws_size,
                              hipStream_t stream)
{
    (void)in_sizes; (void)n_in; (void)out_size; (void)ws_size;
    const float* x       = (const float*)d_in[0];
    const float* wq      = (const float*)d_in[1];
    const float* wk      = (const float*)d_in[2];
    const float* wv      = (const float*)d_in[3];
    const float* wo      = (const float*)d_in[4];
    const float* k_cache = (const float*)d_in[5];
    const float* v_cache = (const float*)d_in[6];
    const int*   start   = (const int*)d_in[7];
    float* ws  = (float*)d_ws;
    float* out = (float*)d_out;

    qkv_rope_kernel<<<768, 256, 0, stream>>>(x, wq, wk, wv, start, ws);
    attn_partial_kernel<<<dim3(NCHUNK, NKV), 256, 0, stream>>>(k_cache, v_cache, start, ws);
    combine_kernel<<<32, 128, 0, stream>>>(ws);
    oproj_kernel<<<512, 256, 0, stream>>>(wo, ws, out);
}
